// GCN_GETS_27393301414248
// MI455X (gfx1250) — matrix operand order repaired and checked by reading, not compiled, not run on hardware
//
#include <hip/hip_runtime.h>
#include <hip/hip_bf16.h>

typedef __attribute__((ext_vector_type(16))) __bf16 v16bf;
typedef __attribute__((ext_vector_type(8)))  float  v8f;

struct bfpair { __bf16 h; __bf16 l; };

__device__ __forceinline__ bfpair splitf(float x) {
    bfpair r;
    r.h = (__bf16)x;
    r.l = (__bf16)(x - (float)r.h);
    return r;
}

// 3-term bf16 product: a*b ~= ah*bh + ah*bl + al*bh  (f32 accumulate)
__device__ __forceinline__ v8f wmma3(v16bf ah, v16bf al, v16bf bh, v16bf bl, v8f acc) {
    acc = __builtin_amdgcn_wmma_f32_16x16x32_bf16(false, ah, false, bh, (short)0, acc, false, false);
    acc = __builtin_amdgcn_wmma_f32_16x16x32_bf16(false, ah, false, bl, (short)0, acc, false, false);
    acc = __builtin_amdgcn_wmma_f32_16x16x32_bf16(false, al, false, bh, (short)0, acc, false, false);
    return acc;
}

// A fragment (16x32 bf16): lane holds K-runs [c1..c1+7] -> slots 0..7,
// [c2..c2+7] -> slots 8..15   (c1 = k0+8g, c2 = k0+16+8g)
__device__ __forceinline__ void load_a(const float* __restrict__ r, int c1, int c2,
                                       v16bf& hi, v16bf& lo) {
#pragma unroll
    for (int p = 0; p < 8; ++p) {
        bfpair t1 = splitf(r[c1 + p]); hi[p]     = t1.h; lo[p]     = t1.l;
        bfpair t2 = splitf(r[c2 + p]); hi[8 + p] = t2.h; lo[8 + p] = t2.l;
    }
}

// ---------------- weight pre-split / pre-pack ----------------
// Packed layout = exact WMMA B-fragment layout: element i decodes as
// i = ((kt*ntiles + nt)*32 + lane)*16 + p ; value = W[(kt*32+(p<8 ? 8g+p : 16+8g+p-8))*C + nt*16+mn]
__global__ __launch_bounds__(256) void pack_w_kernel(const float* __restrict__ W,
                                                     __bf16* __restrict__ hi,
                                                     __bf16* __restrict__ lo,
                                                     int K, int C) {
    int i = blockIdx.x * 256 + threadIdx.x;
    if (i >= K * C) return;
    int p    = i & 15;
    int lane = (i >> 4) & 31;
    int f    = i >> 9;
    int ntiles = C >> 4;
    int kt = f / ntiles, nt = f % ntiles;
    int g = lane >> 4, mn = lane & 15;
    int k = kt * 32 + (p < 8 ? 8 * g + p : 16 + 8 * g + (p - 8));
    int c = nt * 16 + mn;
    bfpair t = splitf(W[(size_t)k * C + c]);
    hi[i] = t.h;
    lo[i] = t.l;
}

// ---------------- utility kernels ----------------

__global__ __launch_bounds__(256) void zero_kernel(float* __restrict__ p, int n) {
    int i = blockIdx.x * 256 + threadIdx.x;
    if (i < n) p[i] = 0.0f;
}

__global__ __launch_bounds__(256) void deg_kernel(const int* __restrict__ src,
                                                  const int* __restrict__ dst,
                                                  float* __restrict__ outdeg,
                                                  float* __restrict__ indeg, int E) {
    int e = blockIdx.x * 256 + threadIdx.x;
    if (e < E) {
        atomicAdd(&outdeg[src[e]], 1.0f);
        atomicAdd(&indeg[dst[e]], 1.0f);
    }
}

__global__ __launch_bounds__(256) void aux_kernel(const float* __restrict__ outdeg,
                                                  const float* __restrict__ indeg,
                                                  float* __restrict__ cs, float* __restrict__ cd,
                                                  int* __restrict__ degidx, int N) {
    int r = blockIdx.x * 256 + threadIdx.x;
    if (r < N) {
        float od = outdeg[r], id = indeg[r];
        cs[r] = rsqrtf(fmaxf(od, 1.0f));
        cd[r] = rsqrtf(fmaxf(id, 1.0f));
        degidx[r] = (int)(od + id);
    }
}

// ---------------- GEMM kernels (WMMA bf16x3, wave = 16 rows x all cols) ----------------

// feat[N,128] = features[N,512] @ proj_W[512,128] + proj_b   (16 ktiles, 8 ntiles)
__global__ __launch_bounds__(256) void proj_kernel(const float* __restrict__ F,
                                                   const v16bf* __restrict__ Bh,
                                                   const v16bf* __restrict__ Bl,
                                                   const float* __restrict__ bias,
                                                   float* __restrict__ feat, int mt_count) {
    const int lane  = threadIdx.x & 31;
    const int mtile = blockIdx.x * 8 + (threadIdx.x >> 5);
    if (mtile >= mt_count) return;             // wave-uniform
    const int g  = lane >> 4;
    const int mn = lane & 15;
    const float* arow = F + (size_t)(mtile * 16 + mn) * 512;
    v8f acc[8] = {};
    for (int kt = 0; kt < 16; ++kt) {
        const int k0 = kt * 32;
        v16bf ah, al;
        load_a(arow, k0 + 8 * g, k0 + 16 + 8 * g, ah, al);
#pragma unroll
        for (int nt = 0; nt < 8; ++nt) {
            const int fi = (kt * 8 + nt) * 32 + lane;
            acc[nt] = wmma3(ah, al, Bh[fi], Bl[fi], acc[nt]);
        }
    }
#pragma unroll
    for (int nt = 0; nt < 8; ++nt) {
        const float bv = bias[nt * 16 + mn];
#pragma unroll
        for (int v = 0; v < 8; ++v)
            feat[(size_t)(mtile * 16 + v + 8 * g) * 128 + nt * 16 + mn] = acc[nt][v] + bv;
    }
}

// h1[N,128] = cs * (concat(logits|feat|deg_emb)[N,256] @ W1[256,128])   (8 ktiles, 8 ntiles)
__global__ __launch_bounds__(256) void gemm1_kernel(const float* __restrict__ logits,
                                                    const float* __restrict__ feat,
                                                    const float* __restrict__ deg_table,
                                                    const int* __restrict__ degidx,
                                                    const float* __restrict__ cs,
                                                    const v16bf* __restrict__ Bh,
                                                    const v16bf* __restrict__ Bl,
                                                    float* __restrict__ h1, int mt_count) {
    const int lane  = threadIdx.x & 31;
    const int mtile = blockIdx.x * 8 + (threadIdx.x >> 5);
    if (mtile >= mt_count) return;
    const int g  = lane >> 4;
    const int mn = lane & 15;
    const int arowi = mtile * 16 + mn;
    const float* aL = logits    + (size_t)arowi * 64;
    const float* aF = feat      + (size_t)arowi * 128;
    const float* aD = deg_table + (size_t)degidx[arowi] * 64;
    v8f acc[8] = {};
#pragma unroll
    for (int kt = 0; kt < 8; ++kt) {
        const float* srcp; int off;
        if (kt < 2)      { srcp = aL; off = kt * 32; }         // K 0..63   : logits
        else if (kt < 6) { srcp = aF; off = kt * 32 - 64; }    // K 64..191 : feat
        else             { srcp = aD; off = kt * 32 - 192; }   // K 192..255: deg_emb
        v16bf ah, al;
        load_a(srcp, off + 8 * g, off + 16 + 8 * g, ah, al);
#pragma unroll
        for (int nt = 0; nt < 8; ++nt) {
            const int fi = (kt * 8 + nt) * 32 + lane;
            acc[nt] = wmma3(ah, al, Bh[fi], Bl[fi], acc[nt]);
        }
    }
#pragma unroll
    for (int nt = 0; nt < 8; ++nt) {
#pragma unroll
        for (int v = 0; v < 8; ++v) {
            int r = mtile * 16 + v + 8 * g;
            h1[(size_t)r * 128 + nt * 16 + mn] = acc[nt][v] * cs[r];
        }
    }
}

// h2[N,64] = cs * ( relu(agg1*cd + b1) @ W2[128,64] )   (4 ktiles, 4 ntiles; conv1 epilogue fused)
__global__ __launch_bounds__(256) void gemm2_kernel(const float* __restrict__ agg1,
                                                    const float* __restrict__ b1,
                                                    const float* __restrict__ cs,
                                                    const float* __restrict__ cd,
                                                    const v16bf* __restrict__ Bh,
                                                    const v16bf* __restrict__ Bl,
                                                    float* __restrict__ h2, int mt_count) {
    const int lane  = threadIdx.x & 31;
    const int mtile = blockIdx.x * 8 + (threadIdx.x >> 5);
    if (mtile >= mt_count) return;
    const int g  = lane >> 4;
    const int mn = lane & 15;
    const int arowi = mtile * 16 + mn;
    const float* ar = agg1 + (size_t)arowi * 128;
    const float cdr = cd[arowi];
    v8f acc[4] = {};
#pragma unroll
    for (int kt = 0; kt < 4; ++kt) {
        const int c1 = kt * 32 + 8 * g;
        const int c2 = kt * 32 + 16 + 8 * g;
        v16bf ah, al;
#pragma unroll
        for (int p = 0; p < 8; ++p) {
            float x1 = fmaxf(ar[c1 + p] * cdr + b1[c1 + p], 0.0f);
            float x2 = fmaxf(ar[c2 + p] * cdr + b1[c2 + p], 0.0f);
            bfpair t1 = splitf(x1); ah[p]     = t1.h; al[p]     = t1.l;
            bfpair t2 = splitf(x2); ah[8 + p] = t2.h; al[8 + p] = t2.l;
        }
#pragma unroll
        for (int nt = 0; nt < 4; ++nt) {
            const int fi = (kt * 4 + nt) * 32 + lane;
            acc[nt] = wmma3(ah, al, Bh[fi], Bl[fi], acc[nt]);
        }
    }
#pragma unroll
    for (int nt = 0; nt < 4; ++nt) {
#pragma unroll
        for (int v = 0; v < 8; ++v) {
            int r = mtile * 16 + v + 8 * g;
            h2[(size_t)r * 64 + nt * 16 + mn] = acc[nt][v] * cs[r];
        }
    }
}

// ---------------- SpMM scatter (edge gather + atomic add) ----------------

template <int LOGC, int NCOL>
__global__ __launch_bounds__(256) void scatter_kernel(const float* __restrict__ h,
                                                      const int* __restrict__ src,
                                                      const int* __restrict__ dst,
                                                      float* __restrict__ agg) {
    unsigned tid = blockIdx.x * 256u + threadIdx.x;
    unsigned e = tid >> LOGC;
    unsigned c = (tid & ((1u << LOGC) - 1u)) * 4u;
    int s = src[e], d = dst[e];
    const float4 v = *(const float4*)(h + (size_t)s * NCOL + c);
    float* o = agg + (size_t)d * NCOL + c;
    atomicAdd(o + 0, v.x);
    atomicAdd(o + 1, v.y);
    atomicAdd(o + 2, v.z);
    atomicAdd(o + 3, v.w);
}

__global__ __launch_bounds__(256) void final_kernel(float* __restrict__ out,
                                                    const float* __restrict__ cd,
                                                    const float* __restrict__ b2) {
    int tid = blockIdx.x * 256 + threadIdx.x;
    int r = tid >> 6, c = tid & 63;
    out[tid] = out[tid] * cd[r] + b2[c];
}

// ---------------- launch ----------------

extern "C" void kernel_launch(void* const* d_in, const int* in_sizes, int n_in,
                              void* d_out, int out_size, void* d_ws, size_t ws_size,
                              hipStream_t stream) {
    const int*   src       = (const int*)d_in[0];
    const int*   dst       = (const int*)d_in[1];
    const float* logits    = (const float*)d_in[2];
    const float* features  = (const float*)d_in[3];
    const float* proj_W    = (const float*)d_in[4];
    const float* proj_b    = (const float*)d_in[5];
    const float* deg_table = (const float*)d_in[6];
    const float* W1        = (const float*)d_in[7];
    const float* b1        = (const float*)d_in[8];
    const float* W2        = (const float*)d_in[9];
    const float* b2        = (const float*)d_in[10];

    const int E = in_sizes[0];
    const int N = in_sizes[2] / 64;   // 100000; divisible by 32
    const int MT = N / 16;            // 6250 row tiles
    const int MB = (MT + 7) / 8;      // blocks of 8 waves (1 M-tile per wave)

    // packed bf16 hi/lo weights first (32B-aligned v16bf loads)
    __bf16* pw_hi = (__bf16*)d_ws;                  // 512*128
    __bf16* pw_lo = pw_hi + 512 * 128;
    __bf16* w1_hi = pw_lo + 512 * 128;              // 256*128
    __bf16* w1_lo = w1_hi + 256 * 128;
    __bf16* w2_hi = w1_lo + 256 * 128;              // 128*64
    __bf16* w2_lo = w2_hi + 128 * 64;
    float*  outdeg = (float*)(w2_lo + 128 * 64);    // [N]
    float*  indeg  = outdeg + N;
    float*  cs     = indeg + N;
    float*  cd     = cs + N;
    int*    degidx = (int*)(cd + N);
    float*  feat   = (float*)(degidx + N);          // [N,128]
    float*  h1     = feat + (size_t)N * 128;        // [N,128]
    float*  agg1   = h1 + (size_t)N * 128;          // [N,128]
    float*  h2     = feat;                          // alias: feat dead after gemm1
    float*  out    = (float*)d_out;                 // [N,64]

    pack_w_kernel<<<(512 * 128) / 256, 256, 0, stream>>>(proj_W, pw_hi, pw_lo, 512, 128);
    pack_w_kernel<<<(256 * 128) / 256, 256, 0, stream>>>(W1, w1_hi, w1_lo, 256, 128);
    pack_w_kernel<<<(128 * 64) / 256, 256, 0, stream>>>(W2, w2_hi, w2_lo, 128, 64);

    zero_kernel<<<(2 * N + 255) / 256, 256, 0, stream>>>(outdeg, 2 * N);
    zero_kernel<<<(N * 128 + 255) / 256, 256, 0, stream>>>(agg1, N * 128);
    zero_kernel<<<(N * 64 + 255) / 256, 256, 0, stream>>>(out, N * 64);

    deg_kernel<<<(E + 255) / 256, 256, 0, stream>>>(src, dst, outdeg, indeg, E);
    aux_kernel<<<(N + 255) / 256, 256, 0, stream>>>(outdeg, indeg, cs, cd, degidx, N);

    proj_kernel<<<MB, 256, 0, stream>>>(features, (const v16bf*)pw_hi, (const v16bf*)pw_lo,
                                        proj_b, feat, MT);
    gemm1_kernel<<<MB, 256, 0, stream>>>(logits, feat, deg_table, degidx, cs,
                                         (const v16bf*)w1_hi, (const v16bf*)w1_lo, h1, MT);
    scatter_kernel<5, 128><<<(unsigned)((size_t)E * 32 / 256), 256, 0, stream>>>(h1, src, dst, agg1);
    gemm2_kernel<<<MB, 256, 0, stream>>>(agg1, b1, cs, cd,
                                         (const v16bf*)w2_hi, (const v16bf*)w2_lo, h2, MT);
    scatter_kernel<4, 64><<<(unsigned)((size_t)E * 16 / 256), 256, 0, stream>>>(h2, src, dst, out);
    final_kernel<<<(N * 64) / 256, 256, 0, stream>>>(out, cd, b2);
}
